// ActionVectorQuantizer_68650757259330
// MI455X (gfx1250) — compile-verified, matrix operand order checked by reading, not run on hardware
//
#include <hip/hip_runtime.h>

// Vector quantizer forward: z [B,256] f32, embedding [4,256] f32.
// out = concat( z_q [B,256] f32 , idx [B] (written as float) ).
//
// Bandwidth-bound (~0.54 GB total @ 23.3 TB/s ~ 23 us). Distances computed as
// score_c = ||e_c||^2 - 2 * (z . e_c); the z.e^T term runs on the matrix pipe
// via v_wmma_f32_16x16x32_bf16 on 16-row tiles (codes padded 4 -> 16 cols).
// z loads and z_q stores are non-temporal (each byte touched exactly once;
// streams exceed the 192 MB L2, so keep them from thrashing it).

#define CODE_DIM 256
#define N_CODES  4

typedef __attribute__((ext_vector_type(16))) __bf16 v16bf;
typedef __attribute__((ext_vector_type(8)))  float  v8f;
typedef __attribute__((ext_vector_type(4)))  float  vf4;

__device__ __forceinline__ v16bf cvt16(const vf4 p0, const vf4 p1,
                                       const vf4 p2, const vf4 p3) {
  v16bf r;
  r[0]  = (__bf16)p0[0]; r[1]  = (__bf16)p0[1]; r[2]  = (__bf16)p0[2]; r[3]  = (__bf16)p0[3];
  r[4]  = (__bf16)p1[0]; r[5]  = (__bf16)p1[1]; r[6]  = (__bf16)p1[2]; r[7]  = (__bf16)p1[3];
  r[8]  = (__bf16)p2[0]; r[9]  = (__bf16)p2[1]; r[10] = (__bf16)p2[2]; r[11] = (__bf16)p2[3];
  r[12] = (__bf16)p3[0]; r[13] = (__bf16)p3[1]; r[14] = (__bf16)p3[2]; r[15] = (__bf16)p3[3];
  return r;
}

__device__ __forceinline__ float sq4(const vf4 p) {
  return p[0]*p[0] + p[1]*p[1] + p[2]*p[2] + p[3]*p[3];
}

__global__ __launch_bounds__(256)
void vq_wmma_kernel(const float* __restrict__ z,
                    const float* __restrict__ emb,
                    float* __restrict__ zq,
                    float* __restrict__ idxOut,
                    int numTiles) {
  // Stage the 4 KB codebook in LDS (f32, exact) for B-preload and output rows.
  __shared__ vf4 semb4[N_CODES * CODE_DIM / 4];
  const int tid = threadIdx.x;
  if (tid < N_CODES * CODE_DIM / 4) semb4[tid] = ((const vf4*)emb)[tid];
  __syncthreads();
  const float* semb = (const float*)semb4;

  const int lane   = tid & 31;
  const int laneM  = lane & 15;   // A: row-in-tile. B/C/D: column (code id).
  const int laneHi = lane >> 4;   // K-half selector for A/B layout.
  const int base   = lane & 16;   // lane group base for score gathering.

  // ---- Preload B = embedding^T (32x16 bf16) for each of the 8 K-steps,
  // ---- accumulating ||e||^2 in f32 along the way. Codes 4..15 are zero.
  v16bf barr[8];
  float eSqPart = 0.0f;
#pragma unroll
  for (int kb = 0; kb < 8; ++kb) {
    vf4 p0 = {0,0,0,0}, p1 = {0,0,0,0}, p2 = {0,0,0,0}, p3 = {0,0,0,0};
    if (laneM < N_CODES) {
      const vf4* er = (const vf4*)(semb + laneM * CODE_DIM);
      const int f0 = (kb * 32 + laneHi * 8) >> 2;       // K = kb*32 + hi*8 + 0..7
      const int f1 = (kb * 32 + 16 + laneHi * 8) >> 2;  // K = kb*32 + 16 + hi*8 + 0..7
      p0 = er[f0]; p1 = er[f0 + 1]; p2 = er[f1]; p3 = er[f1 + 1];
    }
    eSqPart += sq4(p0) + sq4(p1) + sq4(p2) + sq4(p3);
    barr[kb] = cvt16(p0, p1, p2, p3);
  }
  // Each lane holds half of code laneM's dims; combine halves across the wave.
  const float eSq = eSqPart + __shfl_xor(eSqPart, 16, 32);

  const int totalWaves = (gridDim.x * blockDim.x) >> 5;
  const int gwave      = (blockIdx.x * blockDim.x + tid) >> 5;

  for (int tile = gwave; tile < numTiles; tile += totalWaves) {
    const size_t rowA = (size_t)tile * 16 + laneM;
    const vf4* zr     = (const vf4*)(z + rowA * CODE_DIM);

    v8f c = {0,0,0,0,0,0,0,0};
#pragma unroll
    for (int kb = 0; kb < 8; ++kb) {
      const int f0 = (kb * 32 + laneHi * 8) >> 2;
      const int f1 = (kb * 32 + 16 + laneHi * 8) >> 2;
      // Streaming reads: non-temporal (read-once, bypass/deprioritize caching).
      const vf4 p0 = __builtin_nontemporal_load(zr + f0);
      const vf4 p1 = __builtin_nontemporal_load(zr + f0 + 1);
      const vf4 p2 = __builtin_nontemporal_load(zr + f1);
      const vf4 p3 = __builtin_nontemporal_load(zr + f1 + 1);
      const v16bf a = cvt16(p0, p1, p2, p3);
      // D = A(16x32 bf16) x B(32x16 bf16) + C(16x16 f32)
      c = __builtin_amdgcn_wmma_f32_16x16x32_bf16(
              /*neg_a=*/false, a, /*neg_b=*/false, barr[kb],
              /*c_mod=*/(short)0, c, /*reuse_a=*/false, /*reuse_b=*/false);
    }

    // c[r] at this lane = dot(z[tile*16 + r + 8*laneHi], e[laneM]).
#pragma unroll
    for (int r = 0; r < 8; ++r) {
      const float score = eSq - 2.0f * c[r];
      // Row (r + 8*laneHi)'s 4 code scores live in lanes base+0..3.
      const float s0 = __shfl(score, base + 0, 32);
      const float s1 = __shfl(score, base + 1, 32);
      const float s2 = __shfl(score, base + 2, 32);
      const float s3 = __shfl(score, base + 3, 32);
      int q = 0; float best = s0;
      if (s1 < best) { best = s1; q = 1; }
      if (s2 < best) { best = s2; q = 2; }
      if (s3 < best) { best = s3; q = 3; }

      const size_t orow = (size_t)tile * 16 + r + 8 * laneHi;
      // Half-wave (16 lanes) writes one 256-float row: 4 contiguous 256B bursts.
      vf4* dst       = (vf4*)(zq + orow * CODE_DIM);
      const vf4* src = (const vf4*)(semb + q * CODE_DIM);
#pragma unroll
      for (int j = 0; j < 4; ++j) {
        // Write-once stream: non-temporal store.
        __builtin_nontemporal_store(src[j * 16 + laneM], dst + j * 16 + laneM);
      }
      if (laneM == 0) idxOut[orow] = (float)q;  // idx tail stored as float
    }
  }
}

extern "C" void kernel_launch(void* const* d_in, const int* in_sizes, int n_in,
                              void* d_out, int out_size, void* d_ws, size_t ws_size,
                              hipStream_t stream) {
  const float* z   = (const float*)d_in[0];
  const float* emb = (const float*)d_in[1];
  float* out = (float*)d_out;

  const int batch    = in_sizes[0] / CODE_DIM;   // 262144
  const int numTiles = batch / 16;               // 16384 (batch divisible by 16)

  float* zq     = out;
  float* idxOut = out + (size_t)batch * CODE_DIM;

  // 8 waves/block; enough waves in flight for 23.3 TB/s, grid-stride over tiles.
  int blocks = (numTiles + 7) / 8;
  if (blocks > 1024) blocks = 1024;

  vq_wmma_kernel<<<blocks, 256, 0, stream>>>(z, emb, zq, idxOut, numTiles);
}
